// MyRNN_29807073034646
// MI455X (gfx1250) — matrix operand order repaired and checked by reading, not compiled, not run on hardware
//
#include <hip/hip_runtime.h>

// ---------------------------------------------------------------------------
// 2-layer LSTM (B=128, T=80, U=2048, E=100) for MI455X (gfx1250, wave32).
//  * Weights packed once per launch to f16 WMMA B-tiles (96 MB -> L2-resident;
//    each step streams from L2, not HBM @ 23.3 TB/s).
//  * Per step/layer: M=128 x N=8192 GEMM via v_wmma_f32_16x16x32_f16.
//    Tiling: 64 blocks x 8 waves; each wave owns 2 M-tiles x 4 gate-tiles
//    (8 accums) -> 8 WMMA per 12 b128 loads per K-step (0.75 KB/WMMA).
//  * WGP-scope global_prefetch_b8 (inline asm; the builtin can only reach
//    SE/SYS scope which skips the WGP cache) primes L0 for the B stream that
//    4 waves per block re-read.
//  * Fused sigmoid/tanh/cell-state epilogue in registers (z never hits mem).
//  * x@W0 folded into layer-0 K (A=[h0|x_t], B=[U0;W0pad]); layer-1 K=4096.
//  * Ping-pong A buffers kill cross-block h WAR hazards over the 80 steps.
// ---------------------------------------------------------------------------

typedef _Float16 v16h __attribute__((ext_vector_type(16)));
typedef _Float16 v8h  __attribute__((ext_vector_type(8)));
typedef float    v8f  __attribute__((ext_vector_type(8)));

#define BATCH 128
#define SEQ   80
#define EMBD  100
#define UNITS 2048
#define NTOT  (4*UNITS)      // 8192 gate columns (i|f|g|o)
#define K0    2176           // 2048 (h0) + 128 (x padded from 100)
#define K1    4096           // 2048 (h0) + 2048 (h1)
#define NKT0  (K0/32)        // 68 K-tiles
#define NKT1  (K1/32)        // 128 K-tiles
#define NTILES_N 512         // 8192/16

__device__ __forceinline__ float sigmoidf_(float x){ return 1.0f/(1.0f+__expf(-x)); }
__device__ __forceinline__ float tanhf_(float x){ return 2.0f*sigmoidf_(2.0f*x)-1.0f; }

// WGP-scope prefetch (scope 0 = default encoding): pulls into all cache
// levels incl. WGP$, unlike the SE/SYS scopes reachable via __builtin_prefetch.
__device__ __forceinline__ void prefetch_wgp(const void* p){
  asm volatile("global_prefetch_b8 %0, off" :: "v"(p));
}

// ---------------------------------------------------------------- zero scratch
__global__ void zero_ws(unsigned int* __restrict__ p, long n){
  long i = (long)blockIdx.x*blockDim.x + threadIdx.x;
  long s = (long)gridDim.x*blockDim.x;
  for (; i < n; i += s) p[i] = 0u;
}

// ------------------------------------------------------------- weight packing
// Packed B layout: tile(kt,nt) = 512 halves at ((nt*numKt)+kt)*512;
// within a tile, lane l holds halves [l*16 .. l*16+15]:
//   lanes 0-15 : N = nt*16 + l,    K = kt*32 + h       (h = 0..15)
//   lanes 16-31: N = nt*16 + l-16, K = kt*32 + 16 + h
// mode 0: rows 0..2047 = U0, rows 2048..2175 = W0 (e<100) else 0   (layer 0)
// mode 1: rows 0..2047 = W1, rows 2048..4095 = U1                  (layer 1)
__global__ void pack_b(const float* __restrict__ S0, const float* __restrict__ S1,
                       _Float16* __restrict__ dst, int numKt, int mode){
  const long total = (long)numKt * NTILES_N * 512;
  long i = (long)blockIdx.x*blockDim.x + threadIdx.x;
  long stride = (long)gridDim.x*blockDim.x;
  for (; i < total; i += stride){
    long tile = i >> 9;
    int  rem  = (int)(i & 511);
    int  lane = rem >> 4;
    int  h    = rem & 15;
    int  kt   = (int)(tile % numKt);
    long nt   = tile / numKt;
    int  k    = kt*32 + ((lane >= 16) ? 16 : 0) + h;
    int  n    = (int)nt*16 + (lane & 15);
    float v;
    if (k < UNITS) v = S0[(long)k*NTOT + n];
    else {
      int k2 = k - UNITS;
      v = (mode == 0) ? ((k2 < EMBD) ? S1[(long)k2*NTOT + n] : 0.0f)
                      : S1[(long)k2*NTOT + n];
    }
    dst[i] = (_Float16)v;
  }
}

// ----------------------------------------------------- embedding gather (f16)
__global__ void embed_x(const int* __restrict__ tokens, const float* __restrict__ emb,
                        _Float16* __restrict__ a0, int t){
  int i = blockIdx.x*blockDim.x + threadIdx.x;      // 128 rows x 128 padded cols
  if (i >= BATCH*128) return;
  int m = i >> 7, e = i & 127;
  int tok = tokens[m*SEQ + t];
  float v = (e < EMBD) ? emb[(long)tok*EMBD + e] : 0.0f;
  a0[(long)m*K0 + UNITS + e] = (_Float16)v;
}

// ----------------------------------------------------------- fused LSTM layer
// Grid: 64 blocks (32 units each). Block: 256 threads = 8 waves.
// Wave w: mgrp = w&3 -> M rows [32*mgrp, 32*mgrp+32); slice = w>>2 -> 16 units.
// Each wave: 2 M-tiles x 4 gates = 8 WMMA accumulators.
__global__ void __launch_bounds__(256)
lstm_gemm(const _Float16* __restrict__ A, int strideA, int numKt,
          const _Float16* __restrict__ Bp,
          const float* __restrict__ bias,
          float* __restrict__ c,
          _Float16* __restrict__ h0d, int h0stride,
          _Float16* __restrict__ h1d, int h1stride)
{
  const int tid   = threadIdx.x;
  const int w     = tid >> 5;
  const int lane  = tid & 31;
  const int mgrp  = w & 3;                     // 4 groups of 32 rows
  const int slice = w >> 2;                    // 2 unit-slices of 16
  const int u     = blockIdx.x*32 + slice*16 + (lane & 15);

  // A fragments: contiguous K order matching packed B (lane<16: M=lane, K 0-15;
  // lane>=16: M=lane-16, K 16-31).
  const int hiK = (lane >= 16) ? 16 : 0;
  const _Float16* aPtr0 = A + (long)(mgrp*32 +      (lane & 15))*strideA + hiK;
  const _Float16* aPtr1 = A + (long)(mgrp*32 + 16 + (lane & 15))*strideA + hiK;

  const long ntb = (long)blockIdx.x*2 + slice;
  const _Float16* b0p = Bp + ((0*128 + ntb) * numKt)*512 + lane*16;
  const _Float16* b1p = Bp + ((1*128 + ntb) * numKt)*512 + lane*16;
  const _Float16* b2p = Bp + ((2*128 + ntb) * numKt)*512 + lane*16;
  const _Float16* b3p = Bp + ((3*128 + ntb) * numKt)*512 + lane*16;

  v8f ac00 = {}; v8f ac10 = {}; v8f ac20 = {}; v8f ac30 = {};   // M-tile 0
  v8f ac01 = {}; v8f ac11 = {}; v8f ac21 = {}; v8f ac31 = {};   // M-tile 1

  #pragma unroll 2
  for (int kt = 0; kt < numKt; ++kt){
    // WGP-scope prefetch of the B stream ~8 K-tiles (8 KB) ahead
    prefetch_wgp(b0p + (long)(kt + 8)*512);
    prefetch_wgp(b2p + (long)(kt + 8)*512);

    v8h a0l = *(const v8h*)(aPtr0 + kt*32);
    v8h a0h = *(const v8h*)(aPtr0 + kt*32 + 8);
    v8h a1l = *(const v8h*)(aPtr1 + kt*32);
    v8h a1h = *(const v8h*)(aPtr1 + kt*32 + 8);
    v16h a0 = __builtin_shufflevector(a0l, a0h, 0,1,2,3,4,5,6,7,8,9,10,11,12,13,14,15);
    v16h a1 = __builtin_shufflevector(a1l, a1h, 0,1,2,3,4,5,6,7,8,9,10,11,12,13,14,15);

    v16h bf0 = *(const v16h*)(b0p + (long)kt*512);
    v16h bf1 = *(const v16h*)(b1p + (long)kt*512);
    v16h bf2 = *(const v16h*)(b2p + (long)kt*512);
    v16h bf3 = *(const v16h*)(b3p + (long)kt*512);

    ac00 = __builtin_amdgcn_wmma_f32_16x16x32_f16(false, a0, false, bf0, (short)0, ac00, false, false);
    ac01 = __builtin_amdgcn_wmma_f32_16x16x32_f16(false, a1, false, bf0, (short)0, ac01, false, false);
    ac10 = __builtin_amdgcn_wmma_f32_16x16x32_f16(false, a0, false, bf1, (short)0, ac10, false, false);
    ac11 = __builtin_amdgcn_wmma_f32_16x16x32_f16(false, a1, false, bf1, (short)0, ac11, false, false);
    ac20 = __builtin_amdgcn_wmma_f32_16x16x32_f16(false, a0, false, bf2, (short)0, ac20, false, false);
    ac21 = __builtin_amdgcn_wmma_f32_16x16x32_f16(false, a1, false, bf2, (short)0, ac21, false, false);
    ac30 = __builtin_amdgcn_wmma_f32_16x16x32_f16(false, a0, false, bf3, (short)0, ac30, false, false);
    ac31 = __builtin_amdgcn_wmma_f32_16x16x32_f16(false, a1, false, bf3, (short)0, ac31, false, false);
  }

  // Fused gate epilogue. C/D layout: VGPR e, lanes 0-15 -> M=e, lanes 16-31 -> M=8+e.
  const float bi = bias[u], bf_ = bias[UNITS+u], bg = bias[2*UNITS+u], bo = bias[3*UNITS+u];
  const int rEpi = mgrp*32 + ((lane >= 16) ? 8 : 0);
  #pragma unroll
  for (int mt = 0; mt < 2; ++mt){
    const v8f& A0 = mt ? ac01 : ac00;
    const v8f& A1 = mt ? ac11 : ac10;
    const v8f& A2 = mt ? ac21 : ac20;
    const v8f& A3 = mt ? ac31 : ac30;
    #pragma unroll
    for (int e = 0; e < 8; ++e){
      int row = rEpi + mt*16 + e;
      float ig = sigmoidf_(A0[e] + bi);
      float fg = sigmoidf_(A1[e] + bf_);
      float gg = tanhf_   (A2[e] + bg);
      float og = sigmoidf_(A3[e] + bo);
      long ci = (long)row*UNITS + u;
      float cn = fg * c[ci] + ig * gg;
      c[ci] = cn;
      _Float16 hh = (_Float16)(og * tanhf_(cn));
      h0d[(long)row*h0stride + u] = hh;
      if (h1d) h1d[(long)row*h1stride + u] = hh;
    }
  }
}

// ------------------------------------------------------------------ head
__global__ void out_head(const _Float16* __restrict__ h1, const float* __restrict__ Wout,
                         const float* __restrict__ bout, float* __restrict__ out){
  int m = threadIdx.x;
  if (m >= BATCH) return;
  const _Float16* hr = h1 + (long)m*K1;
  float s = 0.0f;
  #pragma unroll 8
  for (int j = 0; j < UNITS; ++j) s += (float)hr[j] * Wout[j];
  out[m] = sigmoidf_(s + bout[0]);
}

// ---------------------------------------------------------------------------
extern "C" void kernel_launch(void* const* d_in, const int* in_sizes, int n_in,
                              void* d_out, int out_size, void* d_ws, size_t ws_size,
                              hipStream_t stream){
  (void)in_sizes; (void)n_in; (void)out_size; (void)ws_size;
  const int*   tokens = (const int*)  d_in[0];
  const float* emb    = (const float*)d_in[1];
  const float* W0     = (const float*)d_in[2];
  const float* U0     = (const float*)d_in[3];
  const float* b0     = (const float*)d_in[4];
  const float* W1     = (const float*)d_in[5];
  const float* U1     = (const float*)d_in[6];
  const float* b1     = (const float*)d_in[7];
  const float* Wout   = (const float*)d_in[8];
  const float* bout   = (const float*)d_in[9];
  float* out = (float*)d_out;

  // Workspace layout (all regions naturally aligned)
  char* ws = (char*)d_ws;
  size_t off = 0;
  _Float16* W0p = (_Float16*)(ws + off); off += (size_t)NKT0*NTILES_N*512*2; // 35.6 MB
  _Float16* W1p = (_Float16*)(ws + off); off += (size_t)NKT1*NTILES_N*512*2; // 64 MB
  size_t zstart = off;
  _Float16* A0b[2];
  A0b[0] = (_Float16*)(ws + off); off += (size_t)BATCH*K0*2;
  A0b[1] = (_Float16*)(ws + off); off += (size_t)BATCH*K0*2;
  _Float16* A1b[2];
  A1b[0] = (_Float16*)(ws + off); off += (size_t)BATCH*K1*2;
  A1b[1] = (_Float16*)(ws + off); off += (size_t)BATCH*K1*2;
  float* c0 = (float*)(ws + off); off += (size_t)BATCH*UNITS*4;
  float* c1 = (float*)(ws + off); off += (size_t)BATCH*UNITS*4;
  size_t zbytes = off - zstart;                       // A0x2 + A1x2 + c0 + c1

  // 1) zero recurrent state + A buffers (deterministic per launch)
  zero_ws<<<2048, 256, 0, stream>>>((unsigned int*)(ws + zstart), (long)(zbytes/4));

  // 2) pack weights to f16 WMMA tiles (one-time per launch; L2-resident after)
  pack_b<<<4096, 256, 0, stream>>>(U0, W0, W0p, NKT0, 0);   // [U0 ; W0pad]
  pack_b<<<8192, 256, 0, stream>>>(W1, U1, W1p, NKT1, 1);   // [W1 ; U1]

  // 3) the 80-step serial chain
  for (int t = 0; t < SEQ; ++t){
    _Float16* A0r = A0b[t & 1];        // holds h0_{t-1}; x_t written below
    _Float16* A0w = A0b[(t + 1) & 1];  // receives h0_t
    _Float16* A1r = A1b[t & 1];        // holds h1_{t-1}; receives h0_t
    _Float16* A1w = A1b[(t + 1) & 1];  // receives h1_t

    embed_x<<<64, 256, 0, stream>>>(tokens, emb, A0r, t);

    // layer 0: z0 = [h0|x_t] @ [U0;W0] ; writes h0_t into next A0 and this A1
    lstm_gemm<<<64, 256, 0, stream>>>(A0r, K0, NKT0, W0p, b0, c0,
                                      A0w, K0, A1r, K1);
    // layer 1: z1 = [h0_t|h1_{t-1}] @ [W1;U1] ; writes h1_t into next A1
    lstm_gemm<<<64, 256, 0, stream>>>(A1r, K1, NKT1, W1p, b1, c1,
                                      A1w + UNITS, K1, ((_Float16*)0), 0);
  }

  // 4) head: h1_final lives in A1b[(79+1)&1] = A1b[0], cols 2048..4095
  out_head<<<1, 128, 0, stream>>>(A1b[0] + UNITS, Wout, bout, out);
}
